// APGCN_47785806135398
// MI455X (gfx1250) — compile-verified, matrix operand order checked by reading, not run on hardware
//
#include <hip/hip_runtime.h>

#define N_NODES 50000
#define N_EDGES 800000
#define F_IN    512
#define HID     64
#define N_CLS   64
#define NITER   10

typedef __bf16 bf16;
typedef bf16  v16bf __attribute__((ext_vector_type(16)));
typedef float v8f   __attribute__((ext_vector_type(8)));

#define KT1 (F_IN / 32)   // 16 k-tiles for GEMM 1
#define KT2 (HID  / 32)   // 2  k-tiles for GEMM 2

// ---------------------------------------------------------------------------
// Fused MLP: local = relu(x @ W1 + b1) @ W2 + b2
// One wave32 per 16-row tile (50000/16 = 3125 tiles exactly).
// W1/W2 staged in LDS in *fragment-major* bf16 layout: each lane's 16
// B-fragment elements are contiguous (32B) -> fragment load = 2x ds_load_b128.
// x streamed fp32 (global b128 clauses + prefetch) -> bf16; WMMA, fp32 acc.
// ---------------------------------------------------------------------------
__launch_bounds__(256)
__global__ void apgcn_mlp_wmma(const float* __restrict__ x,
                               const float* __restrict__ W1,
                               const float* __restrict__ b1,
                               const float* __restrict__ W2,
                               const float* __restrict__ b2,
                               float* __restrict__ local_out) {
  // fragment-major: sW1f[((kt*4 + nt)*32 + lane) * 16 + i]
  __shared__ __attribute__((aligned(32))) bf16 sW1f[KT1 * 4 * 32 * 16]; // 64 KB
  __shared__ __attribute__((aligned(32))) bf16 sW2f[KT2 * 4 * 32 * 16]; //  8 KB
  __shared__ __attribute__((aligned(32))) bf16 sHf[8][KT2 * 32 * 16];   // 16 KB

  const int tid = threadIdx.x;

  // ---- stage W1 into fragment-major bf16 --------------------------------
  for (int g = tid; g < KT1 * 4 * 32; g += 256) {
    const int lg  = g & 31;
    const int nt  = (g >> 5) & 3;
    const int kt  = g >> 7;
    const int kbg = (lg < 16) ? 0 : 8;
    const int n   = nt * 16 + (lg & 15);
    v16bf t;
#pragma unroll
    for (int i = 0; i < 16; ++i) {
      const int k = kt * 32 + kbg + (i < 8 ? i : i + 8);
      t[i] = (bf16)W1[k * HID + n];
    }
    *(v16bf*)&sW1f[g * 16] = t;
  }
  // ---- stage W2 into fragment-major bf16 --------------------------------
  for (int g = tid; g < KT2 * 4 * 32; g += 256) {
    const int lg  = g & 31;
    const int nt  = (g >> 5) & 3;
    const int kt  = g >> 7;
    const int kbg = (lg < 16) ? 0 : 8;
    const int n   = nt * 16 + (lg & 15);
    v16bf t;
#pragma unroll
    for (int i = 0; i < 16; ++i) {
      const int k = kt * 32 + kbg + (i < 8 ? i : i + 8);
      t[i] = (bf16)W2[k * N_CLS + n];
    }
    *(v16bf*)&sW2f[g * 16] = t;
  }
  __syncthreads();

  const int wave = tid >> 5;
  const int lane = tid & 31;
  const int tile = blockIdx.x * 8 + wave;
  if (tile >= N_NODES / 16) return;     // wave-uniform guard (EXEC all-1 for WMMA)

  const int mrow = tile * 16 + (lane & 15);   // A-matrix row for this lane
  const int kb   = (lane < 16) ? 0 : 8;       // lane-group K base (16-bit A layout)
  const int mloc = (lane < 16) ? 0 : 8;       // C/D layout M base for this lane half

  // ---- GEMM 1: h = x @ W1 ------------------------------------------------
  v8f acc[4] = {v8f{}, v8f{}, v8f{}, v8f{}};

#pragma unroll 2
  for (int kt = 0; kt < KT1; ++kt) {
    const float* xr = x + (size_t)mrow * F_IN + kt * 32 + kb;
    __builtin_prefetch(xr + 64, 0, 3);        // global_prefetch_b8: next k-tiles
    float4 l0 = *(const float4*)(xr + 0);
    float4 l1 = *(const float4*)(xr + 4);
    float4 h0 = *(const float4*)(xr + 16);
    float4 h1 = *(const float4*)(xr + 20);
    v16bf a;
    a[0]=(bf16)l0.x; a[1]=(bf16)l0.y; a[2]=(bf16)l0.z; a[3]=(bf16)l0.w;
    a[4]=(bf16)l1.x; a[5]=(bf16)l1.y; a[6]=(bf16)l1.z; a[7]=(bf16)l1.w;
    a[8]=(bf16)h0.x; a[9]=(bf16)h0.y; a[10]=(bf16)h0.z; a[11]=(bf16)h0.w;
    a[12]=(bf16)h1.x; a[13]=(bf16)h1.y; a[14]=(bf16)h1.z; a[15]=(bf16)h1.w;

#pragma unroll
    for (int nt = 0; nt < 4; ++nt) {
      v16bf b = *(const v16bf*)&sW1f[((kt * 4 + nt) * 32 + lane) * 16];
      acc[nt] = __builtin_amdgcn_wmma_f32_16x16x32_bf16(
          false, a, false, b, (short)0, acc[nt], false, false);
    }
  }

  // ---- bias + relu; restage h directly into A-fragment layout in LDS ----
  // h(m, n): kt2 = n>>5, nn = n&31; goes to lane (m + 16*((nn>>3)&1)),
  // element ((nn&7) | ((nn&16)>>1)).
  bf16* hbase = sHf[wave];
#pragma unroll
  for (int nt = 0; nt < 4; ++nt) {
    const int n   = nt * 16 + (lane & 15);
    const int nn  = n & 31;
    const int kt2 = n >> 5;
    const int hi  = (nn >> 3) & 1;
    const int ti  = (nn & 7) | ((nn & 16) >> 1);
    const float bias = b1[n];
#pragma unroll
    for (int r = 0; r < 8; ++r) {
      const float v = fmaxf(acc[nt][r] + bias, 0.0f);
      hbase[((kt2 * 32) + (mloc + r) + 16 * hi) * 16 + ti] = (bf16)v;
    }
  }
  // wave-private LDS RAW: DS ops are in-order within a wave; explicit wait
  // keeps the compiler from hoisting the loads above the stores.
  asm volatile("s_wait_dscnt 0x0" ::: "memory");

  // ---- GEMM 2: local = h @ W2 -------------------------------------------
  v8f acc2[4] = {v8f{}, v8f{}, v8f{}, v8f{}};
#pragma unroll
  for (int kt = 0; kt < KT2; ++kt) {
    v16bf a = *(const v16bf*)&hbase[(kt * 32 + lane) * 16];
#pragma unroll
    for (int nt = 0; nt < 4; ++nt) {
      v16bf b = *(const v16bf*)&sW2f[((kt * 4 + nt) * 32 + lane) * 16];
      acc2[nt] = __builtin_amdgcn_wmma_f32_16x16x32_bf16(
          false, a, false, b, (short)0, acc2[nt], false, false);
    }
  }

#pragma unroll
  for (int nt = 0; nt < 4; ++nt) {
    const int n = nt * 16 + (lane & 15);
    const float bias = b2[n];
#pragma unroll
    for (int r = 0; r < 8; ++r) {
      const int row = tile * 16 + mloc + r;
      local_out[(size_t)row * N_CLS + n] = acc2[nt][r] + bias;
    }
  }
}

// ---------------------------------------------------------------------------
// Graph-propagation helper kernels
// ---------------------------------------------------------------------------
__global__ void apgcn_init(float* xacc, float* deg, float* steps, float* sumh,
                           int* cont) {
  const int i = blockIdx.x * blockDim.x + threadIdx.x;
  if (i < N_NODES * N_CLS) xacc[i] = 0.0f;
  if (i < N_NODES) {
    deg[i] = 1.0f;     // self-loop
    steps[i] = 1.0f;
    sumh[i] = 0.0f;
    cont[i] = 1;
  }
}

__global__ void apgcn_degree(const int* __restrict__ ei, float* deg) {
  const int e = blockIdx.x * blockDim.x + threadIdx.x;
  if (e < N_EDGES) atomicAdd(&deg[ei[e]], 1.0f);   // row = edge_index[0]
}

__global__ void apgcn_rsqrt(const float* __restrict__ deg, float* dis) {
  const int i = blockIdx.x * blockDim.x + threadIdx.x;
  if (i < N_NODES) dis[i] = rsqrtf(deg[i]);        // deg >= 1 always
}

// prop_new seeded with the self-loop term dis[i]^2 * prop[i]
__global__ void apgcn_selfloop(const float* __restrict__ prop,
                               const float* __restrict__ dis,
                               float* __restrict__ pnew) {
  const int i = blockIdx.x * blockDim.x + threadIdx.x;
  if (i >= N_NODES * N_CLS) return;
  const float d = dis[i >> 6];
  pnew[i] = d * d * prop[i];
}

// one thread per (edge, channel); consecutive lanes = consecutive channels.
// prop (12.8 MB) + dis fit in the 192 MB L2, so the fp32 atomics resolve at L2.
__global__ void apgcn_scatter(const int* __restrict__ ei,
                              const float* __restrict__ dis,
                              const float* __restrict__ prop,
                              float* __restrict__ pnew) {
  const int t = blockIdx.x * blockDim.x + threadIdx.x;
  if (t >= N_EDGES * N_CLS) return;
  const int e = t >> 6;
  const int c = t & 63;
  const int r  = ei[e];
  const int cl = ei[N_EDGES + e];
  const float w = dis[r] * dis[cl];
  atomicAdd(&pnew[(size_t)cl * N_CLS + c], w * prop[(size_t)r * N_CLS + c]);
}

// one wave32 per node: halting-unit update
__global__ void apgcn_halt(const float* __restrict__ pnew,
                           const float* __restrict__ hw,
                           const float* __restrict__ hb,
                           float* steps, float* sumh, int* cont,
                           float* p_arr, float* contf) {
  const int wid = (blockIdx.x * blockDim.x + threadIdx.x) >> 5;
  const int lane = threadIdx.x & 31;
  if (wid >= N_NODES) return;
  const float* row = pnew + (size_t)wid * N_CLS;
  float acc = row[lane] * hw[lane] + row[lane + 32] * hw[lane + 32];
#pragma unroll
  for (int off = 16; off > 0; off >>= 1) acc += __shfl_xor(acc, off, 32);
  const float hh = 1.0f / (1.0f + __expf(-(acc + hb[0])));

  float sh = sumh[wid];
  float st = steps[wid];
  const int co = cont[wid];
  const int prob = ((sh + hh) < 0.99f) && co;
  const float pf = prob ? 1.0f : 0.0f;
  st += pf;
  sh += pf * hh;
  const int cond = prob && (st < (float)NITER);
  const float p = cond ? sh : (1.0f - sh);
  if (lane == 0) {
    steps[wid] = st;
    sumh[wid]  = sh;
    cont[wid]  = co && prob;
    p_arr[wid] = p;
    contf[wid] = co ? 1.0f : 0.0f;   // cont BEFORE this step's update
  }
}

__global__ void apgcn_blend(const float* __restrict__ pnew,
                            const float* __restrict__ pold,
                            const float* __restrict__ p_arr,
                            const float* __restrict__ contf,
                            float* __restrict__ xacc) {
  const int i = blockIdx.x * blockDim.x + threadIdx.x;
  if (i >= N_NODES * N_CLS) return;
  const int node = i >> 6;
  const float p = p_arr[node];
  const float cf = contf[node];
  xacc[i] += (pnew[i] * p + pold[i] * (1.0f - p)) * cf;
}

// one wave32 per node: out = log_softmax(xacc/steps); aux outputs appended
__global__ void apgcn_final(const float* __restrict__ xacc,
                            const float* __restrict__ steps,
                            const float* __restrict__ sumh,
                            float* __restrict__ out) {
  const int wid = (blockIdx.x * blockDim.x + threadIdx.x) >> 5;
  const int lane = threadIdx.x & 31;
  if (wid >= N_NODES) return;
  const float inv = 1.0f / steps[wid];
  const float v0 = xacc[(size_t)wid * N_CLS + lane] * inv;
  const float v1 = xacc[(size_t)wid * N_CLS + lane + 32] * inv;
  float m = fmaxf(v0, v1);
#pragma unroll
  for (int off = 16; off > 0; off >>= 1) m = fmaxf(m, __shfl_xor(m, off, 32));
  float s = __expf(v0 - m) + __expf(v1 - m);
#pragma unroll
  for (int off = 16; off > 0; off >>= 1) s += __shfl_xor(s, off, 32);
  const float lse = m + __logf(s);
  out[(size_t)wid * N_CLS + lane]      = v0 - lse;
  out[(size_t)wid * N_CLS + lane + 32] = v1 - lse;
  if (lane == 0) {
    out[(size_t)N_NODES * N_CLS + wid]           = steps[wid];
    out[(size_t)N_NODES * N_CLS + N_NODES + wid] = 1.0f - sumh[wid];
  }
}

// ---------------------------------------------------------------------------
extern "C" void kernel_launch(void* const* d_in, const int* in_sizes, int n_in,
                              void* d_out, int out_size, void* d_ws, size_t ws_size,
                              hipStream_t stream) {
  (void)in_sizes; (void)n_in; (void)out_size; (void)ws_size;

  const float* x  = (const float*)d_in[0];
  const int*   ei = (const int*)d_in[1];
  const float* W1 = (const float*)d_in[2];
  const float* b1 = (const float*)d_in[3];
  const float* W2 = (const float*)d_in[4];
  const float* b2 = (const float*)d_in[5];
  const float* hw = (const float*)d_in[6];
  const float* hb = (const float*)d_in[7];

  float* ws    = (float*)d_ws;
  float* propA = ws;                           // N*64
  float* propB = propA + (size_t)N_NODES * N_CLS;
  float* xacc  = propB + (size_t)N_NODES * N_CLS;
  float* deg   = xacc  + (size_t)N_NODES * N_CLS;
  float* dis   = deg   + N_NODES;
  float* steps = dis   + N_NODES;
  float* sumh  = steps + N_NODES;
  float* p_arr = sumh  + N_NODES;
  float* contf = p_arr + N_NODES;
  int*   cont  = (int*)(contf + N_NODES);

  const int NC = N_NODES * N_CLS;              // 3,200,000
  const int EC = N_EDGES * N_CLS;              // 51,200,000

  apgcn_init<<<(NC + 255) / 256, 256, 0, stream>>>(xacc, deg, steps, sumh, cont);
  apgcn_mlp_wmma<<<(3125 + 7) / 8, 256, 0, stream>>>(x, W1, b1, W2, b2, propA);
  apgcn_degree<<<(N_EDGES + 255) / 256, 256, 0, stream>>>(ei, deg);
  apgcn_rsqrt<<<(N_NODES + 255) / 256, 256, 0, stream>>>(deg, dis);

  float* po = propA;
  float* pn = propB;
  for (int it = 0; it < NITER; ++it) {
    apgcn_selfloop<<<(NC + 255) / 256, 256, 0, stream>>>(po, dis, pn);
    apgcn_scatter<<<(EC + 255) / 256, 256, 0, stream>>>(ei, dis, po, pn);
    apgcn_halt<<<(N_NODES * 32 + 255) / 256, 256, 0, stream>>>(
        pn, hw, hb, steps, sumh, cont, p_arr, contf);
    apgcn_blend<<<(NC + 255) / 256, 256, 0, stream>>>(pn, po, p_arr, contf, xacc);
    float* t = po; po = pn; pn = t;
  }

  apgcn_final<<<(N_NODES * 32 + 255) / 256, 256, 0, stream>>>(
      xacc, steps, sumh, (float*)d_out);
}
